// KernelDensityEstimate_49469433315702
// MI455X (gfx1250) — matrix operand order repaired and checked by reading, not compiled, not run on hardware
//
#include <hip/hip_runtime.h>

typedef __attribute__((ext_vector_type(16))) _Float16 v16h;
typedef __attribute__((ext_vector_type(8)))  _Float16 v8h;
typedef __attribute__((ext_vector_type(8)))  float    v8f;

#define N_TOT 4096
#define M_TOT 128
#define Q_TOT 64
#define D_TOT 128
#define BN    128          // A rows per block
#define LDSW  136          // 128 + 8 f16 pad -> 272B row stride, conflict-free

// dens[n, m] = sum_q exp(-0.5/var * (||a_n||^2 - 2 a_n.b_mq + ||b_mq||^2))
__global__ __launch_bounds__(256)
void kde_dens_kernel(const float* __restrict__ A,
                     const float* __restrict__ B,
                     const float* __restrict__ gv,
                     float* __restrict__ dens) {
  __shared__ _Float16 aT[BN * LDSW];
  __shared__ _Float16 bT[Q_TOT * LDSW];
  __shared__ float a2s[BN];
  __shared__ float b2s[Q_TOT];

  const int t    = threadIdx.x;
  const int wave = t >> 5;
  const int lane = t & 31;
  const int nBlk = blockIdx.x;   // 0..31
  const int m    = blockIdx.y;   // 0..127

  // ---- A tile: 128x128 f32 -> f16 LDS, per-row sum of squares (f32) ----
  {
    const int d = lane * 4;                       // coalesced float4 per lane
    #pragma unroll
    for (int i = 0; i < 16; ++i) {
      const int row  = wave + 8 * i;              // one row per wave-iter
      const int gRow = nBlk * BN + row;
      float4 v = *(const float4*)(A + (size_t)gRow * D_TOT + d);
      float sq = v.x * v.x + v.y * v.y + v.z * v.z + v.w * v.w;
      _Float16* p = &aT[row * LDSW + d];
      p[0] = (_Float16)v.x; p[1] = (_Float16)v.y;
      p[2] = (_Float16)v.z; p[3] = (_Float16)v.w;
      sq += __shfl_xor(sq, 1, 32);
      sq += __shfl_xor(sq, 2, 32);
      sq += __shfl_xor(sq, 4, 32);
      sq += __shfl_xor(sq, 8, 32);
      sq += __shfl_xor(sq, 16, 32);
      if (lane == 0) a2s[row] = sq;
    }
  }
  // ---- B tile for cluster m: 64x128 f32 -> f16 LDS, per-q sum of squares ----
  {
    const float* Bg = B + (size_t)m * Q_TOT * D_TOT;
    const int d = lane * 4;
    #pragma unroll
    for (int i = 0; i < 8; ++i) {
      const int q = wave + 8 * i;                 // 0..63
      float4 v = *(const float4*)(Bg + (size_t)q * D_TOT + d);
      float sq = v.x * v.x + v.y * v.y + v.z * v.z + v.w * v.w;
      _Float16* p = &bT[q * LDSW + d];
      p[0] = (_Float16)v.x; p[1] = (_Float16)v.y;
      p[2] = (_Float16)v.z; p[3] = (_Float16)v.w;
      sq += __shfl_xor(sq, 1, 32);
      sq += __shfl_xor(sq, 2, 32);
      sq += __shfl_xor(sq, 4, 32);
      sq += __shfl_xor(sq, 8, 32);
      sq += __shfl_xor(sq, 16, 32);
      if (lane == 0) b2s[q] = sq;
    }
  }
  __syncthreads();

  // ---- WMMA: each wave owns rows [wave*16, wave*16+16), all 64 q columns ----
  const int row16 = lane & 15;    // A: M row / B: N col / C: N col
  const int half  = lane >> 4;    // K-half selector per ISA VGPR layouts
  const int nBase = wave * 16;

  v8f acc[4];
  #pragma unroll
  for (int qt = 0; qt < 4; ++qt) acc[qt] = (v8f){0.f,0.f,0.f,0.f,0.f,0.f,0.f,0.f};

  #pragma unroll
  for (int kc = 0; kc < 4; ++kc) {
    // A frag (16x32 f16): lane<16 holds K {0..7,16..23}, lane>=16 {8..15,24..31}
    const _Float16* ap = &aT[(nBase + row16) * LDSW + kc * 32 + half * 8];
    v8h alo = *(const v8h*)ap;
    v8h ahi = *(const v8h*)(ap + 16);
    v16h aF = __builtin_shufflevector(alo, ahi, 0,1,2,3,4,5,6,7,8,9,10,11,12,13,14,15);
    #pragma unroll
    for (int qt = 0; qt < 4; ++qt) {
      // B frag (32x16): lane holds col q, K {half*8 + 0..7, 16 + half*8 + 0..7}
      const _Float16* bp = &bT[(qt * 16 + row16) * LDSW + kc * 32 + half * 8];
      v8h blo = *(const v8h*)bp;
      v8h bhi = *(const v8h*)(bp + 16);
      v16h bF = __builtin_shufflevector(blo, bhi, 0,1,2,3,4,5,6,7,8,9,10,11,12,13,14,15);
      acc[qt] = __builtin_amdgcn_wmma_f32_16x16x32_f16(
          false, aF, false, bF, (short)0, acc[qt], false, false);
    }
  }

  // ---- fused epilogue: exp + q reduction ----
  const float c0 = -0.5f / gv[0];
  float a2v[8];
  #pragma unroll
  for (int r = 0; r < 8; ++r) a2v[r] = a2s[nBase + r + half * 8];

  float sacc[8] = {0.f,0.f,0.f,0.f,0.f,0.f,0.f,0.f};
  #pragma unroll
  for (int qt = 0; qt < 4; ++qt) {
    const float b2v = b2s[qt * 16 + row16];
    #pragma unroll
    for (int r = 0; r < 8; ++r) {
      const float x = a2v[r] - 2.0f * acc[qt][r] + b2v;
      sacc[r] += __expf(c0 * x);   // v_exp_f32 (TRANS, co-executes with WMMA)
    }
  }
  #pragma unroll
  for (int r = 0; r < 8; ++r) {
    float s = sacc[r];
    s += __shfl_xor(s, 1, 32);     // masks 1..8 stay inside each 16-lane half,
    s += __shfl_xor(s, 2, 32);     // matching the C-matrix row split (M=r vs r+8)
    s += __shfl_xor(s, 4, 32);
    s += __shfl_xor(s, 8, 32);
    if (row16 == 0) {
      const int n = nBlk * BN + nBase + r + half * 8;
      dens[(size_t)n * M_TOT + m] = s;
    }
  }
}

// prob[n, m] = dens[n, m] / (sum_m dens[n, m] + 1e-10), one wave per row
__global__ __launch_bounds__(256)
void kde_norm_kernel(float* __restrict__ out) {
  const int t    = threadIdx.x;
  const int wave = t >> 5;
  const int lane = t & 31;
  const int n    = blockIdx.x * 8 + wave;
  float4* rowp = (float4*)(out + (size_t)n * M_TOT);
  float4 v = rowp[lane];
  float s = v.x + v.y + v.z + v.w;
  s += __shfl_xor(s, 1, 32);
  s += __shfl_xor(s, 2, 32);
  s += __shfl_xor(s, 4, 32);
  s += __shfl_xor(s, 8, 32);
  s += __shfl_xor(s, 16, 32);
  const float inv = 1.0f / (s + 1e-10f);
  v.x *= inv; v.y *= inv; v.z *= inv; v.w *= inv;
  rowp[lane] = v;
}

extern "C" void kernel_launch(void* const* d_in, const int* in_sizes, int n_in,
                              void* d_out, int out_size, void* d_ws, size_t ws_size,
                              hipStream_t stream) {
  (void)in_sizes; (void)n_in; (void)out_size; (void)d_ws; (void)ws_size;
  const float* A  = (const float*)d_in[0];   // [4096, 128] f32
  const float* B  = (const float*)d_in[1];   // [128, 64, 128] f32
  const float* gv = (const float*)d_in[2];   // [1] f32
  float* out = (float*)d_out;                // [4096, 128] f32

  dim3 grid(N_TOT / BN, M_TOT);
  kde_dens_kernel<<<grid, 256, 0, stream>>>(A, B, gv, out);
  kde_norm_kernel<<<N_TOT / 8, 256, 0, stream>>>(out);
}
